// ContextualizedNN_3161095929969
// MI455X (gfx1250) — matrix operand order repaired and checked by reading, not compiled, not run on hardware
//
#include <hip/hip_runtime.h>
#include <hip/hip_bf16.h>

// ---------------- problem constants ----------------
#define NU 100000
#define NI 50000
#define KK 100      // neighbors / score length
#define DD 64       // embedding dim
#define BB 8192     // batch

#define KP 112      // K padded to 7 M-tiles of 16 (rows)
#define KC 128      // K padded to 4 k-chunks of 32 (WMMA K-dim)

typedef __attribute__((ext_vector_type(16))) __bf16 v16bf;
typedef __attribute__((ext_vector_type(8)))  float  v8f;
typedef __attribute__((ext_vector_type(8)))  unsigned short us8;  // 16B vector
typedef __attribute__((ext_vector_type(2)))  unsigned int  u32x2;

// ---------------- LDS layout (dynamic) ----------------
// S   : bf16 [112][128] row-major        28672 B @ 0       (A for pooling; reused as hA/hB)
// Ep  : bf16 frag-order [4kc][4nt][512]  16384 B @ 28672   (B for pooling; reused as hf f32)
// x   : bf16 [112][128] row-major        28672 B @ 45056
// wstage block (byte-mirrors d_ws):      31620 B @ 73728
//   w1p @+0, w2p @+16384, w3p @+24576, w4p @+28672,
//   b1 @+30720, b2 @+30976, b3 @+31232, b4 @+31360, w5 @+31488, b5 @+31616
// red : f32 [128]  512 B @ 105348
// nb  : i32 [112]  448 B @ 105860
#define SMEM_BYTES   106308
#define WSTAGE_BYTES 31620

__device__ __forceinline__ unsigned short f2bf(float f) {
  return __builtin_bit_cast(unsigned short, (__bf16)f);   // native v_cvt on gfx1250
}

// Fragment-order offset for a B-operand element (k = WMMA K index, n = column).
// tile = (k/32)*ntiles + n/16 ; lane = ((k%32)/16)*16 + n%16 ; elem = k%16
__device__ __forceinline__ int bfrag_off(int k, int n, int ntiles) {
  return (((k >> 5) * ntiles + (n >> 4)) * 32 + (((k >> 4) & 1) << 4) + (n & 15)) * 16 + (k & 15);
}

// A fragment (16-bit 16x32, ISA 7.12.2): per lane two contiguous 16B runs.
__device__ __forceinline__ v16bf load_a_frag(const unsigned short* __restrict__ base,
                                             int stride, int mtile, int kc, int lane) {
  const int m    = mtile * 16 + (lane & 15);
  const int half = lane >> 4;
  const unsigned short* p = base + m * stride + kc * 32 + half * 16;
  union { us8 h[2]; v16bf v; } u;
  u.h[0] = *(const us8*)(p);
  u.h[1] = *(const us8*)(p + 8);
  return u.v;
}

// B fragment from fragment-ordered buffer: 32 contiguous bytes per lane.
__device__ __forceinline__ v16bf load_b_frag(const unsigned short* __restrict__ base,
                                             int ntiles, int kc, int nt, int lane) {
  const unsigned short* p = base + ((kc * ntiles + nt) * 32 + lane) * 16;
  union { us8 h[2]; v16bf v; } u;
  u.h[0] = *(const us8*)(p);
  u.h[1] = *(const us8*)(p + 8);
  return u.v;
}

__device__ __forceinline__ v8f wmma_bf16(v16bf a, v16bf b, v8f c) {
  return __builtin_amdgcn_wmma_f32_16x16x32_bf16(false, a, false, b, (short)0, c, false, false);
}

// =============== prep kernel: pack weights (frag order, bf16) + bias tail into d_ws ===============
__global__ void ctxnn_prep_kernel(const float* __restrict__ w1, const float* __restrict__ w2,
                                  const float* __restrict__ w3, const float* __restrict__ w4,
                                  const float* __restrict__ b1, const float* __restrict__ b2,
                                  const float* __restrict__ b3, const float* __restrict__ b4,
                                  const float* __restrict__ w5, const float* __restrict__ b5,
                                  unsigned short* __restrict__ wp) {
  float* wf = (float*)wp;
  const int i = blockIdx.x * blockDim.x + threadIdx.x;
  if (i < 8192)        { const int o = i >> 7, j = i & 127;               wp[        bfrag_off(j, o, 4)] = f2bf(w1[i]); }
  else if (i < 12288)  { const int t = i - 8192,  o = t >> 6, j = t & 63; wp[ 8192 + bfrag_off(j, o, 4)] = f2bf(w2[t]); }
  else if (i < 14336)  { const int t = i - 12288, o = t >> 6, j = t & 63; wp[12288 + bfrag_off(j, o, 2)] = f2bf(w3[t]); }
  else if (i < 15360)  { const int t = i - 14336, o = t >> 5, j = t & 31; wp[14336 + bfrag_off(j, o, 2)] = f2bf(w4[t]); }
  else if (i < 15424)  { wf[7680 + (i - 15360)] = b1[i - 15360]; }
  else if (i < 15488)  { wf[7744 + (i - 15424)] = b2[i - 15424]; }
  else if (i < 15520)  { wf[7808 + (i - 15488)] = b3[i - 15488]; }
  else if (i < 15552)  { wf[7840 + (i - 15520)] = b4[i - 15520]; }
  else if (i < 15584)  { wf[7872 + (i - 15552)] = w5[i - 15552]; }
  else if (i == 15584) { wf[7904] = b5[0]; }
}

// =============== fused main kernel: one workgroup (7 wave32) per batch element ===============
__global__ void __launch_bounds__(224, 2)
ctxnn_fused_kernel(const float* __restrict__ user_emb, const float* __restrict__ item_emb,
                   const float* __restrict__ user_scr, const float* __restrict__ item_scr,
                   const int* __restrict__ user_idx_t, const int* __restrict__ item_idx_t,
                   const int* __restrict__ user_idxs, const int* __restrict__ item_idxs,
                   const unsigned short* __restrict__ wstage, float* __restrict__ out) {
  extern __shared__ char smem[];
  unsigned short* Sl  = (unsigned short*)(smem);
  unsigned short* Ep  = (unsigned short*)(smem + 28672);
  unsigned short* xl  = (unsigned short*)(smem + 45056);
  unsigned short* w1p = (unsigned short*)(smem + 73728);
  unsigned short* w2p = (unsigned short*)(smem + 90112);
  unsigned short* w3p = (unsigned short*)(smem + 98304);
  unsigned short* w4p = (unsigned short*)(smem + 102400);
  float* b1l = (float*)(smem + 104448);
  float* b2l = (float*)(smem + 104704);
  float* b3l = (float*)(smem + 104960);
  float* b4l = (float*)(smem + 105088);
  float* w5l = (float*)(smem + 105216);
  float* b5l = (float*)(smem + 105344);
  float* red = (float*)(smem + 105348);
  int*   nb  = (int*)  (smem + 105860);
  // aliases (dead regions reused by the MLP)
  unsigned short* hA = Sl;                      // [112][64] bf16 (S region, low half)
  unsigned short* hB = Sl + 7168;               // [112][64] bf16 (S region, high half)
  float*          hf = (float*)(smem + 28672);  // [112][32] f32  (Ep region)

  const int tid  = threadIdx.x;   // 224 threads = 7 wave32
  const int lane = tid & 31;
  const int wave = tid >> 5;      // 0..6 == M-tile
  const int b    = blockIdx.x;

  // ---- flat copy of prepacked weights/biases into LDS (b128 traffic, L2-resident src) ----
  {
    const us8* src = (const us8*)wstage;
    us8* dst = (us8*)(smem + 73728);
    for (int i = tid; i < 1976; i += 224) dst[i] = src[i];      // 31616 B
    if (tid == 0)
      *(unsigned int*)(smem + 73728 + 31616) = ((const unsigned int*)wstage)[7904];
  }
  // zero S + Ep (one contiguous region; padding written once, survives both sides)
  {
    us8 z = {};
    us8* dst = (us8*)smem;
    for (int i = tid; i < 2816; i += 224) dst[i] = z;           // 45056 B
  }
  __syncthreads();

  // ================= scored pooling (user side s=0, item side s=1) =================
#pragma unroll 1
  for (int s = 0; s < 2; ++s) {
    const float* emb   = (s == 0) ? user_emb : item_emb;
    const float* scr   = (s == 0) ? user_scr : item_scr;
    const int*   idx_t = (s == 0) ? user_idx_t : item_idx_t;
    const int    cid   = (s == 0) ? user_idxs[b] : item_idxs[b];

    if (tid < KK) {
      const int id = idx_t[(size_t)cid * KK + tid];
      nb[tid] = id;
      __builtin_prefetch(scr + (size_t)id * KK, 0, 1);   // global_prefetch_b8
      __builtin_prefetch(emb + (size_t)id * DD, 0, 1);
    }
    __syncthreads();

    // row-per-wave gather: emb row (float2/lane) -> Ep frag order; scr row (float4/lane) -> Sl
    const int g0 = ((2 * lane) >> 4) * 512 + ((2 * lane) & 15) * 16;          // d = 2*lane
    const int g1 = ((2 * lane + 1) >> 4) * 512 + ((2 * lane + 1) & 15) * 16;  // d = 2*lane+1
    for (int j = wave; j < KK; j += 7) {
      const int id = nb[j];
      const int f_j = ((j >> 5) << 11) + (((j >> 4) & 1) << 8) + (j & 15);
      const float2 ev = *(const float2*)(emb + (size_t)id * DD + 2 * lane);
      Ep[f_j + g0] = f2bf(ev.x);
      Ep[f_j + g1] = f2bf(ev.y);
      if (lane < 25) {
        const float4 sv = *(const float4*)(scr + (size_t)id * KK + 4 * lane);
        const unsigned int p0 = (unsigned int)f2bf(sv.x) | ((unsigned int)f2bf(sv.y) << 16);
        const unsigned int p1 = (unsigned int)f2bf(sv.z) | ((unsigned int)f2bf(sv.w) << 16);
        *(u32x2*)(Sl + j * KC + 4 * lane) = (u32x2){p0, p1};
      }
    }
    __syncthreads();

    // [112x128] @ [128x64] -> x[:, s*64 : s*64+64]
    for (int nt = 0; nt < 4; ++nt) {
      v8f acc = {};
#pragma unroll
      for (int kc = 0; kc < 4; ++kc)
        acc = wmma_bf16(load_a_frag(Sl, KC, wave, kc, lane),
                        load_b_frag(Ep, 4, kc, nt, lane), acc);
      const int n = nt * 16 + (lane & 15), half = lane >> 4;
#pragma unroll
      for (int r = 0; r < 8; ++r)
        xl[(wave * 16 + half * 8 + r) * KC + s * DD + n] = f2bf(acc[r]);
    }
    __syncthreads();
  }

  // ================= MLP =================
  // L1: [112x128] @ w1^T(128x64) -> hA [112x64], relu
  for (int nt = 0; nt < 4; ++nt) {
    v8f acc = {};
#pragma unroll
    for (int kc = 0; kc < 4; ++kc)
      acc = wmma_bf16(load_a_frag(xl, 128, wave, kc, lane),
                      load_b_frag(w1p, 4, kc, nt, lane), acc);
    const int n = nt * 16 + (lane & 15), half = lane >> 4;
    const float bias = b1l[n];
#pragma unroll
    for (int r = 0; r < 8; ++r) {
      float v = acc[r] + bias; v = v > 0.f ? v : 0.f;
      hA[(wave * 16 + half * 8 + r) * 64 + n] = f2bf(v);
    }
  }
  __syncthreads();

  // L2: [112x64] @ w2^T(64x64) -> hB [112x64], relu
  for (int nt = 0; nt < 4; ++nt) {
    v8f acc = {};
#pragma unroll
    for (int kc = 0; kc < 2; ++kc)
      acc = wmma_bf16(load_a_frag(hA, 64, wave, kc, lane),
                      load_b_frag(w2p, 4, kc, nt, lane), acc);
    const int n = nt * 16 + (lane & 15), half = lane >> 4;
    const float bias = b2l[n];
#pragma unroll
    for (int r = 0; r < 8; ++r) {
      float v = acc[r] + bias; v = v > 0.f ? v : 0.f;
      hB[(wave * 16 + half * 8 + r) * 64 + n] = f2bf(v);
    }
  }
  __syncthreads();

  // L3: [112x64] @ w3^T(64x32) -> hA [112x32], relu
  for (int nt = 0; nt < 2; ++nt) {
    v8f acc = {};
#pragma unroll
    for (int kc = 0; kc < 2; ++kc)
      acc = wmma_bf16(load_a_frag(hB, 64, wave, kc, lane),
                      load_b_frag(w3p, 2, kc, nt, lane), acc);
    const int n = nt * 16 + (lane & 15), half = lane >> 4;
    const float bias = b3l[n];
#pragma unroll
    for (int r = 0; r < 8; ++r) {
      float v = acc[r] + bias; v = v > 0.f ? v : 0.f;
      hA[(wave * 16 + half * 8 + r) * 32 + n] = f2bf(v);
    }
  }
  __syncthreads();

  // L4: [112x32] @ w4^T(32x32) -> hf [112x32] f32, relu
  for (int nt = 0; nt < 2; ++nt) {
    v8f acc = wmma_bf16(load_a_frag(hA, 32, wave, 0, lane),
                        load_b_frag(w4p, 2, 0, nt, lane), (v8f){});
    const int n = nt * 16 + (lane & 15), half = lane >> 4;
    const float bias = b4l[n];
#pragma unroll
    for (int r = 0; r < 8; ++r) {
      float v = acc[r] + bias; v = v > 0.f ? v : 0.f;
      hf[(wave * 16 + half * 8 + r) * 32 + n] = v;
    }
  }
  __syncthreads();

  // L5 + sigmoid + mean over the 100 valid rows (deterministic tree reduce)
  if (tid < 128) {
    float v = 0.f;
    if (tid < KK) {
      float d = b5l[0];
#pragma unroll
      for (int c = 0; c < 32; ++c) d += hf[tid * 32 + c] * w5l[c];
      v = 1.f / (1.f + __expf(-d));
    }
    red[tid] = v;
  }
  __syncthreads();
  if (tid < 32) red[tid] = red[tid] + red[tid + 32] + red[tid + 64] + red[tid + 96];
  __syncthreads();
  if (tid == 0) {
    float acc = 0.f;
#pragma unroll
    for (int i = 0; i < 32; ++i) acc += red[i];
    out[b] = acc * (1.0f / KK);
  }
}

extern "C" void kernel_launch(void* const* d_in, const int* in_sizes, int n_in,
                              void* d_out, int out_size, void* d_ws, size_t ws_size,
                              hipStream_t stream) {
  const float* user_emb   = (const float*)d_in[0];
  const float* item_emb   = (const float*)d_in[1];
  const float* user_scr   = (const float*)d_in[2];
  const float* item_scr   = (const float*)d_in[3];
  const float* w1 = (const float*)d_in[4];
  const float* b1 = (const float*)d_in[5];
  const float* w2 = (const float*)d_in[6];
  const float* b2 = (const float*)d_in[7];
  const float* w3 = (const float*)d_in[8];
  const float* b3 = (const float*)d_in[9];
  const float* w4 = (const float*)d_in[10];
  const float* b4 = (const float*)d_in[11];
  const float* w5 = (const float*)d_in[12];
  const float* b5 = (const float*)d_in[13];
  const int* user_idx_t = (const int*)d_in[14];
  const int* item_idx_t = (const int*)d_in[15];
  const int* user_idxs  = (const int*)d_in[16];
  const int* item_idxs  = (const int*)d_in[17];
  float* out = (float*)d_out;

  (void)in_sizes; (void)n_in; (void)out_size; (void)ws_size;

  // 1) pack weights into d_ws (frag-order bf16 + f32 bias tail)
  ctxnn_prep_kernel<<<dim3(61), dim3(256), 0, stream>>>(
      w1, w2, w3, w4, b1, b2, b3, b4, w5, b5, (unsigned short*)d_ws);

  // 2) fused gather + pooling + MLP
  hipFuncSetAttribute((const void*)ctxnn_fused_kernel,
                      hipFuncAttributeMaxDynamicSharedMemorySize, SMEM_BYTES);
  ctxnn_fused_kernel<<<dim3(BB), dim3(224), SMEM_BYTES, stream>>>(
      user_emb, item_emb, user_scr, item_scr,
      user_idx_t, item_idx_t, user_idxs, item_idxs,
      (const unsigned short*)d_ws, out);
}
